// DFS_17136919511590
// MI455X (gfx1250) — matrix operand order repaired and checked by reading, not compiled, not run on hardware
//
#include <hip/hip_runtime.h>
#include <hip/hip_bf16.h>

typedef __attribute__((ext_vector_type(16))) _Float16 v16h;
typedef __attribute__((ext_vector_type(8)))  float    v8f;
typedef __attribute__((ext_vector_type(2), aligned(4))) float v2fu; // unaligned-safe pair

#define N_   4
#define C_   64
#define H_   192
#define W_   192
#define O_   64
#define KK_  9
#define CIN_ 67
#define OCC_ 18
#define HW_  (H_ * W_)

// ---------------------------------------------------------------------------
// Kernel 1: 3x3 offset conv over concat(x, ref_x) (67 ch) -> 18 offset ch.
// One thread per output pixel, 18 accumulators in registers; weights are
// uniform-indexed (scalar-cache friendly); neighborhood loaded once per ic.
// ---------------------------------------------------------------------------
__global__ __launch_bounds__(256) void dfc_offset_conv(
    const float* __restrict__ x, const float* __restrict__ refx,
    const float* __restrict__ ow, const float* __restrict__ ob,
    float* __restrict__ offout)
{
    int pix = blockIdx.x * 256 + threadIdx.x;
    if (pix >= N_ * H_ * W_) return;
    int w = pix % W_;
    int h = (pix / W_) % H_;
    int n = pix / (W_ * H_);

    int   idx9[9];
    float msk9[9];
#pragma unroll
    for (int i = 0; i < 9; ++i) {
        int yy = h + i / 3 - 1;
        int xx = w + i % 3 - 1;
        bool v = (yy >= 0) && (yy < H_) && (xx >= 0) && (xx < W_);
        int yc = min(max(yy, 0), H_ - 1);
        int xc = min(max(xx, 0), W_ - 1);
        idx9[i] = yc * W_ + xc;
        msk9[i] = v ? 1.0f : 0.0f;
    }

    float acc[OCC_];
#pragma unroll
    for (int oc = 0; oc < OCC_; ++oc) acc[oc] = ob[oc];

    for (int ic = 0; ic < CIN_; ++ic) {
        const float* src = (ic < C_)
            ? (x    + ((size_t)n * C_ + ic)        * HW_)
            : (refx + ((size_t)n * 3  + (ic - C_)) * HW_);
        float v[9];
#pragma unroll
        for (int i = 0; i < 9; ++i) v[i] = msk9[i] * src[idx9[i]];
        const float* wp = ow + (size_t)ic * 9;   // [oc][ic][3][3], oc stride = 67*9
#pragma unroll
        for (int oc = 0; oc < OCC_; ++oc) {
#pragma unroll
            for (int i = 0; i < 9; ++i)
                acc[oc] += wp[(size_t)oc * (CIN_ * 9) + i] * v[i];
        }
    }
#pragma unroll
    for (int oc = 0; oc < OCC_; ++oc)
        offout[((size_t)n * OCC_ + oc) * HW_ + (size_t)h * W_ + w] = acc[oc];
}

// ---------------------------------------------------------------------------
// Kernel 2: fused deformable sampling + WMMA contraction.
// Block = 128 threads = 4 waves. Each wave: 16 contiguous pixels of row h,
// all 64 output channels (4 n-tiles). K = 576 = 9 taps x 64 ch, consumed as
// 18 WMMA K=32 steps. All 9 taps of dc_w staged once as f16 [t][o][c] in
// 72 KB dynamic LDS (CDNA5: 320 KB/WG). Bilinear uses paired b64 row loads
// with boundary handling folded into the pair weights.
// ---------------------------------------------------------------------------
__global__ __launch_bounds__(128) void dfc_deform_wmma(
    const float* __restrict__ x, const float* __restrict__ offset,
    const float* __restrict__ dcw, const float* __restrict__ dcb,
    float* __restrict__ out)
{
    extern __shared__ _Float16 Bt[];    // KK_*O_*C_ = 36864 f16 = 72 KB

    const int tid  = threadIdx.x;
    const int lane = tid & 31;
    const int wv   = tid >> 5;          // wave in block: 0..3
    const int mrow = lane & 15;         // M (pixel) / N (outch) sub-index
    const int hv   = lane >> 4;         // lane half (K-interleave select)

    const int n = blockIdx.z;
    const int h = blockIdx.y;
    const int pixbase = blockIdx.x * 64 + wv * 16;   // W_=192 = 3 tiles of 64
    const int wpix = pixbase + mrow;                 // this lane's sample column

    // ---- stage all taps: Bt[(t*O_ + o)*C_ + c] = (f16) dc_w[o][c][t] ----
    for (int i = tid; i < KK_ * O_ * C_; i += 128) {
        const int t   = i / (O_ * C_);
        const int rem = i - t * (O_ * C_);          // o*C_ + c
        Bt[i] = (_Float16)dcw[(size_t)rem * KK_ + t];
    }
    __syncthreads();

    const float* xn   = x      + (size_t)n * C_   * HW_;
    const float* offn = offset + (size_t)n * OCC_ * HW_;
    const size_t rowoff = (size_t)h * W_ + wpix;

    v8f acc[4] = {};

    for (int t = 0; t < KK_; ++t) {
        // ---- per-pixel bilinear setup (shared across all 64 channels) ----
        const int ky = t / 3 - 1, kx = t % 3 - 1;
        const float dy = offn[(size_t)(2 * t + 0) * HW_ + rowoff];
        const float dx = offn[(size_t)(2 * t + 1) * HW_ + rowoff];
        const float py = (float)(h + ky) + dy;
        const float px = (float)(wpix + kx) + dx;
        const float fy0 = floorf(py), fx0 = floorf(px);
        const float wy1 = py - fy0,  wx1 = px - fx0;
        const int y0 = (int)fy0, x0 = (int)fx0;

        // row weights with validity folded in
        const float ay0 = (y0 >= 0  && y0 <  H_    ) ? (1.0f - wy1) : 0.0f;
        const float ay1 = (y0 >= -1 && y0 <  H_ - 1) ? wy1          : 0.0f;
        const int yb0 = min(max(y0,     0), H_ - 1);
        const int yb1 = min(max(y0 + 1, 0), H_ - 1);

        // x-pair weights: both corners come from one b64 pair at xb, xb+1
        const int   xb  = min(max(x0, 0), W_ - 2);
        const float wx0 = 1.0f - wx1;
        float a0, a1;
        if (x0 >= 0 && x0 <= W_ - 2) { a0 = wx0;  a1 = wx1;  }
        else if (x0 == -1)           { a0 = wx1;  a1 = 0.0f; }  // only x1=0 valid
        else if (x0 == W_ - 1)       { a0 = 0.0f; a1 = wx0;  }  // only x0=W-1 valid
        else                         { a0 = 0.0f; a1 = 0.0f; }  // fully OOB

        const float b00 = ay0 * a0, b01 = ay0 * a1;
        const float b10 = ay1 * a0, b11 = ay1 * a1;
        const float* r0 = xn + (size_t)yb0 * W_ + xb;
        const float* r1 = xn + (size_t)yb1 * W_ + xb;

        // ---- two K=32 channel chunks per tap ----
#pragma unroll
        for (int cc = 0; cc < 2; ++cc) {
            // A fragment: own contiguous K order — lane = M, element e -> K:
            //   K = 16*hv + e (matches B's contiguous order)
            v16h afrag;
#pragma unroll
            for (int e = 0; e < 16; ++e) {
                const int kidx = 16 * hv + e;
                const size_t co = (size_t)(cc * 32 + kidx) * HW_;
                const v2fu q0 = *(const v2fu*)(r0 + co);   // global_load_b64
                const v2fu q1 = *(const v2fu*)(r1 + co);
                const float s = b00 * q0.x + b01 * q0.y
                              + b10 * q1.x + b11 * q1.y;
                afrag[e] = (_Float16)s;
            }
            // B fragments: lane N = mrow, K = 16*hv + e (contiguous 32B in LDS)
#pragma unroll
            for (int nt = 0; nt < 4; ++nt) {
                const v16h bfrag = *(const v16h*)
                    &Bt[((t * O_) + nt * 16 + mrow) * C_ + cc * 32 + 16 * hv];
                acc[nt] = __builtin_amdgcn_wmma_f32_16x16x32_f16(
                    /*neg_a=*/false, afrag, /*neg_b=*/false, bfrag,
                    /*c_mod=*/(short)0, acc[nt],
                    /*reuse_a=*/false, /*reuse_b=*/false);
            }
        }
    }

    // ---- store: D layout — lane N = lane%16, M = vgpr + 8*(lane/16) ----
    // per lane/n-tile: 8 contiguous floats -> merged global_store_b128 x2
#pragma unroll
    for (int nt = 0; nt < 4; ++nt) {
        const int o = nt * 16 + mrow;
        const float bias = dcb[o];
        float* orow = out + (((size_t)n * O_ + o) * H_ + h) * W_ + pixbase + 8 * hv;
#pragma unroll
        for (int r = 0; r < 8; ++r)
            orow[r] = acc[nt][r] + bias;
    }
}

// ---------------------------------------------------------------------------
// Launcher. Workspace layout: [0, N*18*H*W floats) = offset map (~10.1 MB).
// ---------------------------------------------------------------------------
extern "C" void kernel_launch(void* const* d_in, const int* in_sizes, int n_in,
                              void* d_out, int out_size, void* d_ws, size_t ws_size,
                              hipStream_t stream)
{
    (void)in_sizes; (void)n_in; (void)out_size; (void)ws_size;
    const float* x     = (const float*)d_in[0];
    const float* ref_x = (const float*)d_in[1];
    const float* off_w = (const float*)d_in[2];
    const float* off_b = (const float*)d_in[3];
    const float* dc_w  = (const float*)d_in[4];
    const float* dc_b  = (const float*)d_in[5];
    float* out    = (float*)d_out;
    float* offbuf = (float*)d_ws;     // N_*OCC_*HW_ floats

    const int npix = N_ * H_ * W_;
    dfc_offset_conv<<<(npix + 255) / 256, 256, 0, stream>>>(
        x, ref_x, off_w, off_b, offbuf);

    dim3 grid(W_ / 64, H_, N_);       // (3, 192, 4) blocks of 4 waves
    const size_t ldsB = (size_t)KK_ * O_ * C_ * sizeof(_Float16);   // 73728 B
    dfc_deform_wmma<<<grid, 128, ldsB, stream>>>(
        x, offbuf, dc_w, dc_b, out);
}
